// Block_25589415149601
// MI455X (gfx1250) — matrix operand order repaired and checked by reading, not compiled, not run on hardware
//
#include <hip/hip_runtime.h>
#include <hip/hip_bf16.h>
#include <cstdint>

// ---------------- problem constants ----------------
constexpr int B_   = 4;
constexpr int T_   = 2048;
constexpr int D_   = 1024;
constexpr int NH_  = 8;
constexpr int HD_  = 128;
constexpr int HKV_ = 4;          // NH/2
constexpr int MM_  = 4;
constexpr int MROWS = B_ * T_;   // 8192
constexpr float LN_SCALE  = 0.28867513459481287f;  // 1/sqrt(12)
constexpr float EPS_RMS   = 1e-6f;
constexpr float EPS_HEAD  = 1.1920929e-7f;         // finfo(f32).eps
constexpr float ATT_SCALE = 0.08838834764831845f;  // 1/sqrt(128)

typedef __bf16 bf16_t;
typedef bf16_t v16bf __attribute__((ext_vector_type(16)));
typedef float  v8f   __attribute__((ext_vector_type(8)));

// ---------------- WMMA helpers ----------------
__device__ __forceinline__ v8f wmma_bf16(v16bf a, v16bf b, v8f c) {
  // D = A(16x32 bf16) * B(32x16 bf16) + C(16x16 f32)
  return __builtin_amdgcn_wmma_f32_16x16x32_bf16(
      /*neg_a=*/false, a, /*neg_b=*/false, b,
      /*c_mod=*/(short)0, c, /*reuse_a=*/false, /*reuse_b=*/false);
}

// A-matrix 16x32 bf16 VGPR layout (ISA 7.12.2): lane%16 = row M,
// dword j holds K pair; K = ((j>=4)?16:0) + 8*(lane/16) + 2*(j&3)
__device__ __forceinline__ int kmapA(int j, int half) {
  return ((j & 4) ? 16 : 0) + half * 8 + ((j & 3) << 1);
}
// B-matrix 32x16 bf16: lane%16 = column N,
// dword j holds K pair; K = ((j>=4)?16:0) + 8*(lane/16) + 2*(j&3)
__device__ __forceinline__ int kmapB(int j, int half) {
  return ((j & 4) ? 16 : 0) + half * 8 + ((j & 3) << 1);
}

// Load A fragment from row-major [M x K] source (row index = lane%16).
__device__ __forceinline__ v16bf loadA(const bf16_t* __restrict__ src,
                                       int ld, int row0, int k0, int lane) {
  const int half = lane >> 4;
  const bf16_t* p = src + (size_t)(row0 + (lane & 15)) * ld + k0;
  v16bf f;
#pragma unroll
  for (int j = 0; j < 8; ++j) {
    int k = kmapA(j, half);
    f[2 * j]     = p[k];
    f[2 * j + 1] = p[k + 1];
  }
  return f;
}

// Load B fragment (B[k][n]) from row-major [N x K] source (i.e. B^T stored,
// as with weight matrices W[out][in]); row index = lane%16 = n.
__device__ __forceinline__ v16bf loadB(const bf16_t* __restrict__ src,
                                       int ld, int n0, int k0, int lane) {
  const int half = lane >> 4;
  const bf16_t* p = src + (size_t)(n0 + (lane & 15)) * ld + k0;
  v16bf f;
#pragma unroll
  for (int j = 0; j < 8; ++j) {
    int k = kmapB(j, half);
    f[2 * j]     = p[k];
    f[2 * j + 1] = p[k + 1];
  }
  return f;
}

__device__ __forceinline__ float wave_sum(float v) {
#pragma unroll
  for (int off = 16; off > 0; off >>= 1) v += __shfl_xor(v, off, 32);
  return v;
}

// ---------------- CDNA5 async global->LDS copy (ASYNCcnt path) -----------
__device__ __forceinline__ void async_ld_b128(uint32_t lds_byte_off,
                                              const bf16_t* g) {
  // GLOBAL_LOAD_ASYNC_TO_LDS_B128 (GV mode): vdst = LDS byte offset VGPR,
  // vaddr = 64-bit global address. Tracked by ASYNCcnt.
  asm volatile("global_load_async_to_lds_b128 %0, %1, off"
               :: "v"(lds_byte_off), "v"(g)
               : "memory");
}
__device__ __forceinline__ void wait_async0() {
  asm volatile("s_wait_asynccnt 0x0" ::: "memory");
}
__device__ __forceinline__ void wait_async4() {
  asm volatile("s_wait_asynccnt 0x4" ::: "memory");
}

// ---------------- elementwise kernels ----------------
__global__ void cast_bf16_kernel(const float* __restrict__ in,
                                 bf16_t* __restrict__ out, int n) {
  int i = blockIdx.x * blockDim.x + threadIdx.x;
  if (i < n) out[i] = (bf16_t)in[i];
}

// rmsnorm per row of [rows, D_] + scale, emit bf16
__global__ void rmsnorm_kernel(const float* __restrict__ x,
                               bf16_t* __restrict__ out) {
  const int row = blockIdx.x;
  const float* xr = x + (size_t)row * D_;
  bf16_t* orow    = out + (size_t)row * D_;
  float ss = 0.f;
  for (int i = threadIdx.x; i < D_; i += 256) { float v = xr[i]; ss += v * v; }
  __shared__ float red[256];
  red[threadIdx.x] = ss;
  __syncthreads();
#pragma unroll
  for (int s = 128; s > 0; s >>= 1) {
    if ((int)threadIdx.x < s) red[threadIdx.x] += red[threadIdx.x + s];
    __syncthreads();
  }
  const float r = rsqrtf(red[0] * (1.0f / D_) + EPS_RMS) * LN_SCALE;
  for (int i = threadIdx.x; i < D_; i += 256) orow[i] = (bf16_t)(xr[i] * r);
}

// ---------------- bf16 WMMA GEMM: C[M,N] = A[M,K] * W[N,K]^T --------------
// Block tile 128x128 (8 waves, wave tile 32x64). A/W tiles double-buffered
// in LDS, filled with async global->LDS b128 copies, consumed via ds_load.
// EPI 0: C=f32 store.  EPI 1: squared leaky-relu (slope 0.5) -> bf16 outb.  EPI 2: res+scale*acc
template <int EPI>
__global__ __launch_bounds__(256) void gemm_bf16_kernel(
    const bf16_t* __restrict__ A, const bf16_t* __restrict__ W,
    float* __restrict__ C, const float* __restrict__ res,
    const float* __restrict__ scale, bf16_t* __restrict__ outb,
    int M, int N, int K) {
  __shared__ bf16_t Ab[2][128][32];
  __shared__ bf16_t Wb[2][128][32];

  const int tid  = threadIdx.x;
  const int lane = tid & 31;
  const int wv   = tid >> 5;
  const int half = lane >> 4, col = lane & 15;
  const int wm = wv >> 1;            // 0..3  (M sub-tile)
  const int wn = wv & 1;             // 0..1  (N sub-tile)
  const int n0 = blockIdx.x * 128;
  const int m0 = blockIdx.y * 128;

  // Each thread async-copies 2x b128 of the A tile and 2x b128 of the W tile.
  const int tr = tid >> 1;           // tile row 0..127
  const int cb = (tid & 1) * 32;     // byte offset within 64B row (0 / 32)
  const bf16_t* gA = A + (size_t)(m0 + tr) * K + (cb >> 1);
  const bf16_t* gW = W + (size_t)(n0 + tr) * K + (cb >> 1);

  auto stage = [&](int kk, int buf) {
    const bf16_t* ga = gA + kk;
    const bf16_t* gw = gW + kk;
    uint32_t la = (uint32_t)(size_t)&Ab[buf][tr][0] + cb;
    uint32_t lw = (uint32_t)(size_t)&Wb[buf][tr][0] + cb;
    async_ld_b128(la,      ga);
    async_ld_b128(la + 16, ga + 8);
    async_ld_b128(lw,      gw);
    async_ld_b128(lw + 16, gw + 8);
  };

  v8f acc[2][4] = {};
  const int nks = K / 32;
  stage(0, 0);
  for (int ks = 0; ks < nks; ++ks) {
    const int buf = ks & 1;
    if (ks + 1 < nks) {
      stage((ks + 1) * 32, buf ^ 1);  // prefetch next tile while waiting
      wait_async4();                  // previous 4 async ops (this tile) done
    } else {
      wait_async0();
    }
    __syncthreads();
    v16bf a0 = loadA(&Ab[buf][0][0], 32, wm * 32,      0, lane);
    v16bf a1 = loadA(&Ab[buf][0][0], 32, wm * 32 + 16, 0, lane);
#pragma unroll
    for (int nt = 0; nt < 4; ++nt) {
      v16bf bw = loadB(&Wb[buf][0][0], 32, wn * 64 + nt * 16, 0, lane);
      acc[0][nt] = wmma_bf16(a0, bw, acc[0][nt]);
      acc[1][nt] = wmma_bf16(a1, bw, acc[1][nt]);
    }
    __syncthreads();
  }

#pragma unroll
  for (int r = 0; r < 2; ++r) {
    const int mbase = m0 + wm * 32 + r * 16 + half * 8;
#pragma unroll
    for (int nt = 0; nt < 4; ++nt) {
      const int n = n0 + wn * 64 + nt * 16 + col;
#pragma unroll
      for (int i = 0; i < 8; ++i) {
        const size_t idx = (size_t)(mbase + i) * N + n;
        const float v = acc[r][nt][i];
        if (EPI == 0) {
          C[idx] = v;
        } else if (EPI == 1) {
          float t = v >= 0.f ? v : 0.5f * v;
          outb[idx] = (bf16_t)(t * t);
        } else {
          C[idx] = res[idx] + scale[n] * v;
        }
      }
    }
  }
}

// ---------------- head rmsnorm + q-gain + partial rope + rearrange --------
// qf [b,t,NH,HD] ; kf,vf [b,t,HKV,HD]  (f32 gemm outputs)
// -> qb [b,NH,T,HD] bf16 ; kb [b,HKV,T,HD] bf16 ; vtb [b,HKV,HD,T] bf16
__global__ __launch_bounds__(384) void qkv_rope_kernel(
    const float* __restrict__ qf, const float* __restrict__ kf,
    const float* __restrict__ vf, const float* __restrict__ cosT,
    const float* __restrict__ sinT, const float* __restrict__ qgain,
    bf16_t* __restrict__ qb, bf16_t* __restrict__ kb,
    bf16_t* __restrict__ vtb) {
  const int bt = blockIdx.x;
  const int b = bt / T_, t = bt % T_;
  const int w = threadIdx.x >> 5, lane = threadIdx.x & 31;
  float f[4];

  if (w < NH_) {  // q heads
    const float* src = qf + ((size_t)bt * NH_ + w) * HD_;
#pragma unroll
    for (int j = 0; j < 4; ++j) f[j] = src[lane * 4 + j];
    float ss = f[0]*f[0] + f[1]*f[1] + f[2]*f[2] + f[3]*f[3];
    ss = wave_sum(ss);
    const float r = rsqrtf(ss * (1.0f / HD_) + EPS_HEAD) * qgain[w];
#pragma unroll
    for (int j = 0; j < 4; ++j) f[j] *= r;
    if (lane < 4) {  // rope dims 0..15 live in lanes 0..3 (pairs in-lane)
#pragma unroll
      for (int p = 0; p < 2; ++p) {
        const int i = lane * 2 + p;
        const float c = cosT[t * 8 + i], s = sinT[t * 8 + i];
        const float x1 = f[2 * p], x2 = f[2 * p + 1];
        f[2 * p]     = x1 * c - x2 * s;
        f[2 * p + 1] = x2 * c + x1 * s;
      }
    }
    bf16_t* dst = qb + (((size_t)b * NH_ + w) * T_ + t) * HD_;
#pragma unroll
    for (int j = 0; j < 4; ++j) dst[lane * 4 + j] = (bf16_t)f[j];
  } else {  // kv heads
    const int hk = w - NH_;
    const float* src = kf + ((size_t)bt * HKV_ + hk) * HD_;
#pragma unroll
    for (int j = 0; j < 4; ++j) f[j] = src[lane * 4 + j];
    float ss = f[0]*f[0] + f[1]*f[1] + f[2]*f[2] + f[3]*f[3];
    ss = wave_sum(ss);
    const float r = rsqrtf(ss * (1.0f / HD_) + EPS_HEAD);
#pragma unroll
    for (int j = 0; j < 4; ++j) f[j] *= r;
    if (lane < 4) {
#pragma unroll
      for (int p = 0; p < 2; ++p) {
        const int i = lane * 2 + p;
        const float c = cosT[t * 8 + i], s = sinT[t * 8 + i];
        const float x1 = f[2 * p], x2 = f[2 * p + 1];
        f[2 * p]     = x1 * c - x2 * s;
        f[2 * p + 1] = x2 * c + x1 * s;
      }
    }
    bf16_t* dst = kb + (((size_t)b * HKV_ + hk) * T_ + t) * HD_;
#pragma unroll
    for (int j = 0; j < 4; ++j) dst[lane * 4 + j] = (bf16_t)f[j];
    // v: transposed store [b,hk,hd,t]
    const float* vs = vf + ((size_t)bt * HKV_ + hk) * HD_;
#pragma unroll
    for (int j = 0; j < 4; ++j) {
      vtb[(((size_t)b * HKV_ + hk) * HD_ + lane * 4 + j) * T_ + t] =
          (bf16_t)vs[lane * 4 + j];
    }
  }
}

// ---------------- flash attention: one wave per 16 query rows -------------
__global__ __launch_bounds__(32) void flash_attn_kernel(
    const bf16_t* __restrict__ Q, const bf16_t* __restrict__ Kc,
    const bf16_t* __restrict__ Vt, float* __restrict__ Y) {
  const int qb  = blockIdx.x;              // 16-row query block
  const int bh  = blockIdx.y;              // b*NH + h
  const int b   = bh / NH_, h = bh % NH_;
  const int kvh = h >> 1;                  // group = 2
  const int lane = threadIdx.x;
  const int half = lane >> 4, col = lane & 15;

  const bf16_t* Qp = Q  + ((size_t)(b * NH_  + h)   * T_)  * HD_;
  const bf16_t* Kp = Kc + ((size_t)(b * HKV_ + kvh) * T_)  * HD_;
  const bf16_t* Vp = Vt + ((size_t)(b * HKV_ + kvh) * HD_) * T_;

  v16bf qf[4];
#pragma unroll
  for (int d = 0; d < 4; ++d) qf[d] = loadA(Qp, HD_, qb * 16, d * 32, lane);

  v8f acc[8] = {};
  float mrow[8], lrow[8], alpha[8];
#pragma unroll
  for (int i = 0; i < 8; ++i) { mrow[i] = -3.0e38f; lrow[i] = 0.f; }

  __shared__ float pbuf[16][33];  // P tile staging (C-layout -> A-layout)

  const int nkb = (qb * 16 + 16 + 31) / 32;
  for (int kb = 0; kb < nkb; ++kb) {
    const int k0 = kb * 32;
    v8f s0 = {}, s1 = {};
#pragma unroll
    for (int d = 0; d < 4; ++d) {
      v16bf b0 = loadB(Kp, HD_, k0, d * 32, lane);        // B[d][key]
      s0 = wmma_bf16(qf[d], b0, s0);
    }
#pragma unroll
    for (int d = 0; d < 4; ++d) {
      v16bf b1 = loadB(Kp, HD_, k0 + 16, d * 32, lane);
      s1 = wmma_bf16(qf[d], b1, s1);
    }
    // scale + causal mask + online softmax (row lives in one 16-lane half)
#pragma unroll
    for (int i = 0; i < 8; ++i) {
      const int m = qb * 16 + i + half * 8;
      float v0 = s0[i] * ATT_SCALE; if (k0 + col      > m) v0 = -3.0e38f;
      float v1 = s1[i] * ATT_SCALE; if (k0 + 16 + col > m) v1 = -3.0e38f;
      float rm = fmaxf(v0, v1);
      rm = fmaxf(rm, __shfl_xor(rm, 1, 32));
      rm = fmaxf(rm, __shfl_xor(rm, 2, 32));
      rm = fmaxf(rm, __shfl_xor(rm, 4, 32));
      rm = fmaxf(rm, __shfl_xor(rm, 8, 32));
      const float mn = fmaxf(mrow[i], rm);
      const float al = __expf(mrow[i] - mn);
      const float p0 = __expf(v0 - mn);
      const float p1 = __expf(v1 - mn);
      float rs = p0 + p1;
      rs += __shfl_xor(rs, 1, 32);
      rs += __shfl_xor(rs, 2, 32);
      rs += __shfl_xor(rs, 4, 32);
      rs += __shfl_xor(rs, 8, 32);
      mrow[i] = mn;
      lrow[i] = lrow[i] * al + rs;
      alpha[i] = al;
      s0[i] = p0;
      s1[i] = p1;
    }
#pragma unroll
    for (int nt = 0; nt < 8; ++nt)
#pragma unroll
      for (int i = 0; i < 8; ++i) acc[nt][i] *= alpha[i];

    // transpose P 16x32 through LDS into A-fragment layout
#pragma unroll
    for (int i = 0; i < 8; ++i) {
      pbuf[i + half * 8][col]      = s0[i];
      pbuf[i + half * 8][col + 16] = s1[i];
    }
    __syncthreads();
    v16bf pf;
#pragma unroll
    for (int j = 0; j < 8; ++j) {
      const int k = kmapA(j, half);
      pf[2 * j]     = (bf16_t)pbuf[col][k];
      pf[2 * j + 1] = (bf16_t)pbuf[col][k + 1];
    }
    __syncthreads();
    // P(16x32) * V(32xHD): Vt row-major [HD,T] -> B fragment per 16-col tile
#pragma unroll
    for (int nt = 0; nt < 8; ++nt) {
      v16bf vb = loadB(Vp, T_, nt * 16, k0, lane);
      acc[nt] = wmma_bf16(pf, vb, acc[nt]);
    }
  }
  // write Y [b, t, h, hd] f32 (normalized by row sum)
#pragma unroll
  for (int nt = 0; nt < 8; ++nt)
#pragma unroll
    for (int i = 0; i < 8; ++i) {
      const int m = qb * 16 + i + half * 8;
      Y[(((size_t)b * T_ + m) * NH_ + h) * HD_ + nt * 16 + col] =
          acc[nt][i] / lrow[i];
    }
}

// ---------------- v-projection removal: y2 = y - (y.vn)vn -> bf16 --------
__global__ __launch_bounds__(256) void projrm_kernel(
    const float* __restrict__ Y, const float* __restrict__ vf,
    bf16_t* __restrict__ y2b) {
  const int bt = blockIdx.x;
  const int h = threadIdx.x >> 5, lane = threadIdx.x & 31;
  const int kvh = h >> 1;
  const float* y = Y  + ((size_t)bt * NH_  + h)   * HD_;
  const float* v = vf + ((size_t)bt * HKV_ + kvh) * HD_;
  float yv[4], vv[4], sv = 0.f, sy = 0.f;
#pragma unroll
  for (int j = 0; j < 4; ++j) {
    vv[j] = v[lane * 4 + j];
    yv[j] = y[lane * 4 + j];
    sv += vv[j] * vv[j];
    sy += yv[j] * vv[j];
  }
  sv = wave_sum(sv);
  sy = wave_sum(sy);
  const float mx = fmaxf(sqrtf(sv), 1e-12f);
  const float c = sy / (mx * mx);
  bf16_t* o = y2b + (size_t)bt * D_ + h * HD_;
#pragma unroll
  for (int j = 0; j < 4; ++j) o[lane * 4 + j] = (bf16_t)(yv[j] - c * vv[j]);
}

// ---------------- host-side orchestration ----------------
static inline size_t alignup(size_t x) { return (x + 255) & ~(size_t)255; }

extern "C" void kernel_launch(void* const* d_in, const int* in_sizes, int n_in,
                              void* d_out, int out_size, void* d_ws, size_t ws_size,
                              hipStream_t stream) {
  (void)in_sizes; (void)n_in; (void)out_size; (void)ws_size;
  const float* x        = (const float*)d_in[0];
  const float* rope_cos = (const float*)d_in[1];
  const float* rope_sin = (const float*)d_in[2];
  const float* wq       = (const float*)d_in[3];
  const float* wk       = (const float*)d_in[4];
  const float* wv       = (const float*)d_in[5];
  const float* wo       = (const float*)d_in[6];
  const float* wfc      = (const float*)d_in[7];
  const float* wproj    = (const float*)d_in[8];
  const float* attn_sc  = (const float*)d_in[9];
  const float* mlp_sc   = (const float*)d_in[10];
  const float* q_gain   = (const float*)d_in[11];
  float* out = (float*)d_out;

  // workspace carve-up
  char* p = (char*)d_ws;
  auto take = [&](size_t bytes) { char* r = p; p += alignup(bytes); return r; };
  bf16_t* wq_b  = (bf16_t*)take((size_t)D_ * D_ * 2);
  bf16_t* wk_b  = (bf16_t*)take((size_t)(D_/2) * D_ * 2);
  bf16_t* wv_b  = (bf16_t*)take((size_t)(D_/2) * D_ * 2);
  bf16_t* wo_b  = (bf16_t*)take((size_t)D_ * D_ * 2);
  bf16_t* wfc_b = (bf16_t*)take((size_t)(D_*MM_) * D_ * 2);
  bf16_t* wpr_b = (bf16_t*)take((size_t)D_ * (D_*MM_) * 2);
  bf16_t* h_b   = (bf16_t*)take((size_t)MROWS * D_ * 2);
  float*  q_f   = (float* )take((size_t)MROWS * D_ * 4);
  float*  k_f   = (float* )take((size_t)MROWS * (D_/2) * 4);
  float*  v_f   = (float* )take((size_t)MROWS * (D_/2) * 4);
  bf16_t* qb    = (bf16_t*)take((size_t)MROWS * D_ * 2);
  bf16_t* kb    = (bf16_t*)take((size_t)MROWS * (D_/2) * 2);
  bf16_t* vtb   = (bf16_t*)take((size_t)MROWS * (D_/2) * 2);
  float*  Y     = (float* )take((size_t)MROWS * D_ * 4);
  bf16_t* y2b   = (bf16_t*)take((size_t)MROWS * D_ * 2);
  float*  x1    = (float* )take((size_t)MROWS * D_ * 4);
  bf16_t* h2b   = (bf16_t*)take((size_t)MROWS * D_ * 2);
  bf16_t* act_b = (bf16_t*)take((size_t)MROWS * (D_*MM_) * 2);

  auto cast = [&](const float* src, bf16_t* dst, int n) {
    cast_bf16_kernel<<<(n + 255) / 256, 256, 0, stream>>>(src, dst, n);
  };
  cast(wq,    wq_b,  D_ * D_);
  cast(wk,    wk_b,  (D_/2) * D_);
  cast(wv,    wv_b,  (D_/2) * D_);
  cast(wo,    wo_b,  D_ * D_);
  cast(wfc,   wfc_b, (D_*MM_) * D_);
  cast(wproj, wpr_b, D_ * (D_*MM_));

  // h = rmsnorm(x) * LN_SCALE
  rmsnorm_kernel<<<MROWS, 256, 0, stream>>>(x, h_b);

  // q,k,v GEMMs (block tile 128x128)
  gemm_bf16_kernel<0><<<dim3(D_/128,     MROWS/128), 256, 0, stream>>>(
      h_b, wq_b, q_f, nullptr, nullptr, nullptr, MROWS, D_,   D_);
  gemm_bf16_kernel<0><<<dim3((D_/2)/128, MROWS/128), 256, 0, stream>>>(
      h_b, wk_b, k_f, nullptr, nullptr, nullptr, MROWS, D_/2, D_);
  gemm_bf16_kernel<0><<<dim3((D_/2)/128, MROWS/128), 256, 0, stream>>>(
      h_b, wv_b, v_f, nullptr, nullptr, nullptr, MROWS, D_/2, D_);

  // head rmsnorm + gain + rope + rearrange
  qkv_rope_kernel<<<MROWS, 384, 0, stream>>>(q_f, k_f, v_f, rope_cos, rope_sin,
                                             q_gain, qb, kb, vtb);

  // flash attention
  flash_attn_kernel<<<dim3(T_/16, B_*NH_), 32, 0, stream>>>(qb, kb, vtb, Y);

  // remove projection onto normalized v
  projrm_kernel<<<MROWS, 256, 0, stream>>>(Y, v_f, y2b);

  // x1 = x + attn_scale * (y2 @ wo.T)
  gemm_bf16_kernel<2><<<dim3(D_/128, MROWS/128), 256, 0, stream>>>(
      y2b, wo_b, x1, x, attn_sc, nullptr, MROWS, D_, D_);

  // h2 = rmsnorm(x1) * LN_SCALE
  rmsnorm_kernel<<<MROWS, 256, 0, stream>>>(x1, h2b);

  // act = leaky_relu(h2 @ wfc.T, 0.5)^2  (bf16)
  gemm_bf16_kernel<1><<<dim3((D_*MM_)/128, MROWS/128), 256, 0, stream>>>(
      h2b, wfc_b, nullptr, nullptr, nullptr, act_b, MROWS, D_*MM_, D_);

  // out = x1 + mlp_scale * (act @ wproj.T)
  gemm_bf16_kernel<2><<<dim3(D_/128, MROWS/128), 256, 0, stream>>>(
      act_b, wpr_b, out, x1, mlp_sc, nullptr, MROWS, D_, D_*MM_);
}
